// AtomAttentionEncoder_30305289240866
// MI455X (gfx1250) — compile-verified, matrix operand order checked by reading, not run on hardware
//
#include <hip/hip_runtime.h>
#include <hip/hip_bf16.h>
#include <stdint.h>

// ---------------------------------------------------------------------------
// CDNA5 (gfx1250) WMMA types
// ---------------------------------------------------------------------------
typedef __attribute__((ext_vector_type(16))) __bf16 v16bf;
typedef __attribute__((ext_vector_type(8)))  float  v8f;

union FragBF { v16bf v; unsigned int u[8]; };

// native bf16 conversions: let the backend pick v_cvt (RNE) for gfx1250
__device__ __forceinline__ unsigned short f2bf(float f) {
    union { __bf16 h; unsigned short s; } c;
    c.h = (__bf16)f;
    return c.s;
}
__device__ __forceinline__ unsigned int pk2bf(float a, float b) {
    union { __bf16 h[2]; unsigned int u; } c;
    c.h[0] = (__bf16)a;
    c.h[1] = (__bf16)b;
    return c.u;
}

__device__ __forceinline__ v8f vzero8() {
    v8f z;
#pragma unroll
    for (int i = 0; i < 8; ++i) z[i] = 0.0f;
    return z;
}

// K mapping for 16-bit A/B WMMA fragments (ISA 7.12.2): per lane-group g,
// VGPR v holds the bf16 pair (kmap, kmap+1).
__device__ __forceinline__ int wmma_kmap(int v, int g) {
    return ((v < 4) ? (2 * v) : (16 + 2 * (v - 4))) + 8 * g;
}

// ---------------------------------------------------------------------------
// Generic tiled GEMM: C[M,N] = act(A[M,K] @ B[K,N] + bias[N] (+ resid[M,N]))
// bf16 operands staged through LDS (batched b128 loads, packed b64 stores),
// f32 WMMA accumulate.  Tile 128x64, BK=32, 256 threads = 8 wave32 waves.
// Requires M % 128 == 0, K % 32 == 0, leading dims % 4 == 0 (true below).
// ---------------------------------------------------------------------------
#define GBM 128
#define GBN 64
#define GBK 32
#define GLD (GBK + 8)   // LDS row stride in bf16 (even; *2 bytes = 80)

__global__ __launch_bounds__(256)
void gemm_bf16_wmma(const float* __restrict__ A, int lda, long long sAb,
                    const float* __restrict__ B, int ldb, long long sBb, int transB,
                    float* __restrict__ C, int ldc, long long sCb,
                    const float* __restrict__ bias,
                    const float* __restrict__ resid, int ldr,
                    int M, int N, int K, int act)
{
    __shared__ unsigned short As[GBM][GLD];
    __shared__ unsigned short Bs[GBN][GLD];

    const int tid  = threadIdx.x;
    const int lane = tid & 31;
    const int wave = tid >> 5;
    const int wm   = wave & 3;    // wave row (0..3) -> 32 rows each
    const int wn   = wave >> 2;   // wave col (0..1) -> 32 cols each
    const int g    = lane >> 4;
    const int l16  = lane & 15;

    const int tileM = blockIdx.y * GBM;
    const int tileN = blockIdx.x * GBN;

    A += (long long)blockIdx.z * sAb;
    B += (long long)blockIdx.z * sBb;
    C += (long long)blockIdx.z * sCb;

    int kmap[8];
#pragma unroll
    for (int v = 0; v < 8; ++v) kmap[v] = wmma_kmap(v, g);

    v8f acc[2][2];
#pragma unroll
    for (int i = 0; i < 2; ++i)
#pragma unroll
        for (int j = 0; j < 2; ++j) acc[i][j] = vzero8();

    for (int k0 = 0; k0 < K; k0 += GBK) {
        // ---- load phase: issue all global b128 loads back-to-back ----
        float4 av[4];
#pragma unroll
        for (int i = 0; i < 4; ++i) {
            int e4 = tid + 256 * i;            // float4 index (0..1023)
            int r  = e4 >> 3, cq = (e4 & 7) << 2;
            av[i] = *reinterpret_cast<const float4*>(
                        &A[(tileM + r) * lda + (k0 + cq)]);
        }
        float4 bt[2];                           // transB path
        float  bp[8];                           // non-trans path
        if (!transB) {
#pragma unroll
            for (int i = 0; i < 4; ++i) {
                int p = tid + 256 * i;          // k-pair index (0..1023)
                int n = p & 63, k = (p >> 6) << 1;
                bp[2 * i] = 0.0f; bp[2 * i + 1] = 0.0f;
                if (tileN + n < N) {
                    bp[2 * i]     = B[(k0 + k)     * ldb + (tileN + n)];
                    bp[2 * i + 1] = B[(k0 + k + 1) * ldb + (tileN + n)];
                }
            }
        } else {
#pragma unroll
            for (int i = 0; i < 2; ++i) {
                int e4 = tid + 256 * i;         // float4 index (0..511)
                int n  = e4 >> 3, kq = (e4 & 7) << 2;
                bt[i].x = bt[i].y = bt[i].z = bt[i].w = 0.0f;
                if (tileN + n < N)
                    bt[i] = *reinterpret_cast<const float4*>(
                                &B[(tileN + n) * ldb + (k0 + kq)]);
            }
        }

        // ---- convert + store phase ----
#pragma unroll
        for (int i = 0; i < 4; ++i) {
            int e4 = tid + 256 * i;
            int r  = e4 >> 3, cq = (e4 & 7) << 2;
            uint2 p; p.x = pk2bf(av[i].x, av[i].y); p.y = pk2bf(av[i].z, av[i].w);
            *reinterpret_cast<uint2*>(&As[r][cq]) = p;
        }
        if (!transB) {
#pragma unroll
            for (int i = 0; i < 4; ++i) {
                int p = tid + 256 * i;
                int n = p & 63, k = (p >> 6) << 1;
                *reinterpret_cast<unsigned int*>(&Bs[n][k]) =
                    pk2bf(bp[2 * i], bp[2 * i + 1]);
            }
        } else {
#pragma unroll
            for (int i = 0; i < 2; ++i) {
                int e4 = tid + 256 * i;
                int n  = e4 >> 3, kq = (e4 & 7) << 2;
                uint2 p; p.x = pk2bf(bt[i].x, bt[i].y); p.y = pk2bf(bt[i].z, bt[i].w);
                *reinterpret_cast<uint2*>(&Bs[n][kq]) = p;
            }
        }
        // prefetch next A tile while we compute on this one
        if (k0 + GBK < K) {
            int r = tid >> 1, c = (tid & 1) << 4;
            __builtin_prefetch(&A[(tileM + r) * lda + (k0 + GBK) + c], 0, 3);
        }
        __syncthreads();

        // ---- 2x2 WMMA fragments per wave ----
#pragma unroll
        for (int fm = 0; fm < 2; ++fm) {
            FragBF a;
            const int ar = wm * 32 + fm * 16 + l16;
#pragma unroll
            for (int v = 0; v < 8; ++v)
                a.u[v] = *reinterpret_cast<const unsigned int*>(&As[ar][kmap[v]]);
#pragma unroll
            for (int fn = 0; fn < 2; ++fn) {
                FragBF b;
                const int bn = wn * 32 + fn * 16 + l16;
#pragma unroll
                for (int v = 0; v < 8; ++v)
                    b.u[v] = *reinterpret_cast<const unsigned int*>(&Bs[bn][kmap[v]]);
                acc[fm][fn] = __builtin_amdgcn_wmma_f32_16x16x32_bf16(
                    false, a.v, false, b.v, (short)0, acc[fm][fn], false, false);
            }
        }
        __syncthreads();
    }

    // ---- epilogue: bias / residual / activation ----
#pragma unroll
    for (int fm = 0; fm < 2; ++fm) {
#pragma unroll
        for (int fn = 0; fn < 2; ++fn) {
            const int n = tileN + wn * 32 + fn * 16 + l16;
            if (n < N) {
                const float bv = bias ? bias[n] : 0.0f;
#pragma unroll
                for (int v = 0; v < 8; ++v) {
                    const int m = tileM + wm * 32 + fm * 16 + 8 * g + v;
                    float val = acc[fm][fn][v] + bv;
                    if (resid) val += resid[m * ldr + n];
                    if (act == 1)      val = fmaxf(val, 0.0f);
                    else if (act == 2) val = val / (1.0f + __expf(-val));  // SiLU
                    C[m * ldc + n] = val;
                }
            }
        }
    }
}

// ---------------------------------------------------------------------------
// pf = relu((pair_in + ph_i + pw_j) @ m1 + mb1) @ m2 + mb2     [N*N, 16]
// one thread per pair element; 16x16 weights cached in LDS; float4 I/O.
// ---------------------------------------------------------------------------
__global__ __launch_bounds__(256)
void pair_feat_kernel(const float* __restrict__ pair_in,
                      const float* __restrict__ ph, const float* __restrict__ pw,
                      const float* __restrict__ m1, const float* __restrict__ mb1,
                      const float* __restrict__ m2, const float* __restrict__ mb2,
                      float* __restrict__ pf, int N)
{
    __shared__ float sm1[256], sm2[256], sb1[16], sb2[16];
    const int tid = threadIdx.x;
    sm1[tid] = m1[tid];
    sm2[tid] = m2[tid];
    if (tid < 16) { sb1[tid] = mb1[tid]; sb2[tid] = mb2[tid]; }
    __syncthreads();

    const int e = blockIdx.x * 256 + tid;     // < N*N = 262144
    const int i = e / N, j = e % N;

    float x[16], h[16];
#pragma unroll
    for (int q = 0; q < 4; ++q) {
        float4 a = *reinterpret_cast<const float4*>(&pair_in[e * 16 + 4 * q]);
        float4 b = *reinterpret_cast<const float4*>(&ph[i * 16 + 4 * q]);
        float4 c = *reinterpret_cast<const float4*>(&pw[j * 16 + 4 * q]);
        x[4 * q + 0] = a.x + b.x + c.x;
        x[4 * q + 1] = a.y + b.y + c.y;
        x[4 * q + 2] = a.z + b.z + c.z;
        x[4 * q + 3] = a.w + b.w + c.w;
    }
#pragma unroll
    for (int f = 0; f < 16; ++f) {
        float s = sb1[f];
#pragma unroll
        for (int t = 0; t < 16; ++t) s += x[t] * sm1[t * 16 + f];
        h[f] = fmaxf(s, 0.0f);
    }
#pragma unroll
    for (int f = 0; f < 16; ++f) {
        float s = sb2[f];
#pragma unroll
        for (int t = 0; t < 16; ++t) s += h[t] * sm2[t * 16 + f];
        x[f] = s;   // reuse as output staging
    }
#pragma unroll
    for (int q = 0; q < 4; ++q) {
        float4 o; o.x = x[4 * q]; o.y = x[4 * q + 1]; o.z = x[4 * q + 2]; o.w = x[4 * q + 3];
        *reinterpret_cast<float4*>(&pf[e * 16 + 4 * q]) = o;
    }
}

// helper: stage 16 pf rows (K=16, padded to 32) into LDS bf16
__device__ __forceinline__ void stage_pf16(const float* __restrict__ pf, int r0,
                                           unsigned short As[16][40], int tid)
{
    if (tid < 64) {
        int r = tid >> 2, cq = (tid & 3) << 2;
        float4 v = *reinterpret_cast<const float4*>(&pf[(r0 + r) * 16 + cq]);
        uint2 p; p.x = pk2bf(v.x, v.y); p.y = pk2bf(v.z, v.w);
        *reinterpret_cast<uint2*>(&As[r][cq])      = p;
        uint2 z; z.x = 0u; z.y = 0u;
        *reinterpret_cast<uint2*>(&As[r][cq + 16]) = z;   // K pad 16 -> 32
    }
}

// ---------------------------------------------------------------------------
// bias[r] = relu(pf[r,:16] @ w1[16,768] + b1) @ w2[768] + b2
// 16 pair-rows per block; WMMA stage-1; per-lane weighted partials reduced
// through LDS float atomics (ds_add_f32).
// ---------------------------------------------------------------------------
__global__ __launch_bounds__(256)
void pair_bias_kernel(const float* __restrict__ pf,
                      const float* __restrict__ w1, const float* __restrict__ b1,
                      const float* __restrict__ w2, const float* __restrict__ b2,
                      float* __restrict__ out)
{
    __shared__ unsigned short As[16][40];
    __shared__ float red[16];

    const int tid = threadIdx.x, lane = tid & 31, wave = tid >> 5;
    const int g = lane >> 4, l16 = lane & 15;
    const int r0 = blockIdx.x * 16;

    stage_pf16(pf, r0, As, tid);
    if (tid < 16) red[tid] = 0.0f;
    __syncthreads();

    int kmap[8];
#pragma unroll
    for (int v = 0; v < 8; ++v) kmap[v] = wmma_kmap(v, g);

    FragBF a;
#pragma unroll
    for (int v = 0; v < 8; ++v)
        a.u[v] = *reinterpret_cast<const unsigned int*>(&As[l16][kmap[v]]);

    float part[8];
#pragma unroll
    for (int v = 0; v < 8; ++v) part[v] = 0.0f;

    for (int j = 0; j < 6; ++j) {                  // 8 waves * 6 = 48 chunks = 768
        const int f0 = (wave * 6 + j) * 16;
        const int n  = f0 + l16;
        FragBF b;
#pragma unroll
        for (int v = 0; v < 8; ++v) {
            int k = kmap[v];
            unsigned int lo = (k     < 16) ? (unsigned int)f2bf(w1[k * 768 + n])       : 0u;
            unsigned int hi = (k + 1 < 16) ? (unsigned int)f2bf(w1[(k + 1) * 768 + n]) : 0u;
            b.u[v] = lo | (hi << 16);
        }
        v8f h = __builtin_amdgcn_wmma_f32_16x16x32_bf16(
            false, a.v, false, b.v, (short)0, vzero8(), false, false);
        const float bb = b1[n], ww = w2[n];
#pragma unroll
        for (int v = 0; v < 8; ++v)
            part[v] += fmaxf(h[v] + bb, 0.0f) * ww;
    }
#pragma unroll
    for (int v = 0; v < 8; ++v) atomicAdd(&red[8 * g + v], part[v]);
    __syncthreads();
    if (tid < 16) out[r0 + tid] = red[tid] + b2[0];
}

// ---------------------------------------------------------------------------
// token_pair[r,:16] = relu(pf[r,:16] @ w1[16,768] + b1) @ w2[768,16] + b2
// WMMA for both stages; hidden (16x768) held in LDS as bf16.
// ---------------------------------------------------------------------------
__global__ __launch_bounds__(256)
void token_pair_kernel(const float* __restrict__ pf,
                       const float* __restrict__ w1, const float* __restrict__ b1,
                       const float* __restrict__ w2, const float* __restrict__ b2,
                       float* __restrict__ out)
{
    __shared__ unsigned short As[16][40];
    __shared__ unsigned short Hs[16][776];     // 768 + pad
    __shared__ float red[16][16];

    const int tid = threadIdx.x, lane = tid & 31, wave = tid >> 5;
    const int g = lane >> 4, l16 = lane & 15;
    const int r0 = blockIdx.x * 16;

    stage_pf16(pf, r0, As, tid);
    { int r = tid >> 4, c = tid & 15; red[r][c] = 0.0f; }
    __syncthreads();

    int kmap[8];
#pragma unroll
    for (int v = 0; v < 8; ++v) kmap[v] = wmma_kmap(v, g);

    FragBF a;
#pragma unroll
    for (int v = 0; v < 8; ++v)
        a.u[v] = *reinterpret_cast<const unsigned int*>(&As[l16][kmap[v]]);

    // stage 1: hidden = relu(A @ w1 + b1) -> LDS bf16
    for (int j = 0; j < 6; ++j) {
        const int f0 = (wave * 6 + j) * 16;
        const int n  = f0 + l16;
        FragBF b;
#pragma unroll
        for (int v = 0; v < 8; ++v) {
            int k = kmap[v];
            unsigned int lo = (k     < 16) ? (unsigned int)f2bf(w1[k * 768 + n])       : 0u;
            unsigned int hi = (k + 1 < 16) ? (unsigned int)f2bf(w1[(k + 1) * 768 + n]) : 0u;
            b.u[v] = lo | (hi << 16);
        }
        v8f h = __builtin_amdgcn_wmma_f32_16x16x32_bf16(
            false, a.v, false, b.v, (short)0, vzero8(), false, false);
        const float bb = b1[n];
#pragma unroll
        for (int v = 0; v < 8; ++v)
            Hs[8 * g + v][n] = f2bf(fmaxf(h[v] + bb, 0.0f));
    }
    __syncthreads();

    // stage 2: out = hidden @ w2 ; K = 768 = 24 chunks of 32, 3 per wave
    v8f acc = vzero8();
    for (int c = 0; c < 3; ++c) {
        const int k0 = (wave + 8 * c) * 32;
        FragBF ha, hb;
#pragma unroll
        for (int v = 0; v < 8; ++v)
            ha.u[v] = *reinterpret_cast<const unsigned int*>(&Hs[l16][k0 + kmap[v]]);
#pragma unroll
        for (int v = 0; v < 8; ++v) {
            int kk = k0 + kmap[v];
            hb.u[v] = pk2bf(w2[kk * 16 + l16], w2[(kk + 1) * 16 + l16]);
        }
        acc = __builtin_amdgcn_wmma_f32_16x16x32_bf16(
            false, ha.v, false, hb.v, (short)0, acc, false, false);
    }
#pragma unroll
    for (int v = 0; v < 8; ++v) atomicAdd(&red[8 * g + v][l16], acc[v]);
    __syncthreads();

    { int r = tid >> 4, c = tid & 15;
      out[(r0 + r) * 16 + c] = red[r][c] + b2[c]; }
}

// ---------------------------------------------------------------------------
// LayerNorm over D=768 per row
// ---------------------------------------------------------------------------
__global__ __launch_bounds__(256)
void layernorm_kernel(const float* __restrict__ x, const float* __restrict__ gam,
                      const float* __restrict__ bet, float* __restrict__ y, int D)
{
    __shared__ float red[256];
    const int row = blockIdx.x, tid = threadIdx.x;
    const float* xr = x + row * D;
    float* yr = y + row * D;

    float s = 0.0f;
    for (int c = tid; c < D; c += 256) s += xr[c];
    red[tid] = s; __syncthreads();
    for (int o = 128; o > 0; o >>= 1) { if (tid < o) red[tid] += red[tid + o]; __syncthreads(); }
    const float mean = red[0] / D; __syncthreads();

    float v = 0.0f;
    for (int c = tid; c < D; c += 256) { float d = xr[c] - mean; v += d * d; }
    red[tid] = v; __syncthreads();
    for (int o = 128; o > 0; o >>= 1) { if (tid < o) red[tid] += red[tid + o]; __syncthreads(); }
    const float inv = rsqrtf(red[0] / D + 1e-5f);

    for (int c = tid; c < D; c += 256)
        yr[c] = (xr[c] - mean) * inv * gam[c] + bet[c];
}

// ---------------------------------------------------------------------------
// softmax over keys with scale, additive pair bias and mask.  S: [H, N, N]
// ---------------------------------------------------------------------------
__global__ __launch_bounds__(256)
void softmax_bias_kernel(float* __restrict__ S, const float* __restrict__ bias,
                         const unsigned char* __restrict__ mask1,
                         const unsigned char* __restrict__ maskp,
                         int N, float scale)
{
    __shared__ float red[256];
    const int row = blockIdx.x, head = blockIdx.y, tid = threadIdx.x;
    float* srow = S + (head * N + row) * N;
    const float* brow = bias + row * N;
    const bool qm = maskp[row] != 0;
    const float NEG = -3.402823466e38f;

    float vals[2], vmax = NEG;
#pragma unroll
    for (int i = 0; i < 2; ++i) {
        int c = tid + 256 * i;
        float s = srow[c] * scale + brow[c];
        s = (qm && mask1[c] != 0) ? s : NEG;
        vals[i] = s; vmax = fmaxf(vmax, s);
    }
    red[tid] = vmax; __syncthreads();
    for (int o = 128; o > 0; o >>= 1) { if (tid < o) red[tid] = fmaxf(red[tid], red[tid + o]); __syncthreads(); }
    const float rmax = red[0]; __syncthreads();

    float sum = 0.0f;
#pragma unroll
    for (int i = 0; i < 2; ++i) { vals[i] = __expf(vals[i] - rmax); sum += vals[i]; }
    red[tid] = sum; __syncthreads();
    for (int o = 128; o > 0; o >>= 1) { if (tid < o) red[tid] += red[tid + o]; __syncthreads(); }
    const float inv = 1.0f / red[0];
#pragma unroll
    for (int i = 0; i < 2; ++i) srow[tid + 256 * i] = vals[i] * inv;
}

__global__ __launch_bounds__(256)
void copy_kernel(const float* __restrict__ src, float* __restrict__ dst, int n)
{
    int i = blockIdx.x * 256 + threadIdx.x;
    if (i < n) dst[i] = src[i];
}

// ---------------------------------------------------------------------------
// host side
// ---------------------------------------------------------------------------
static inline void launch_gemm(hipStream_t s,
    const float* A, int lda, long long sAb,
    const float* B, int ldb, long long sBb, int transB,
    float* C, int ldc, long long sCb,
    const float* bias, const float* resid, int ldr,
    int M, int N, int K, int act, int batch)
{
    dim3 grid((N + GBN - 1) / GBN, M / GBM, batch);
    gemm_bf16_wmma<<<grid, 256, 0, s>>>(A, lda, sAb, B, ldb, sBb, transB,
                                        C, ldc, sCb, bias, resid, ldr,
                                        M, N, K, act);
}

extern "C" void kernel_launch(void* const* d_in, const int* in_sizes, int n_in,
                              void* d_out, int out_size, void* d_ws, size_t ws_size,
                              hipStream_t stream)
{
    constexpr int N = 512, D = 768, FF = 3072, H = 12, TS = 384, NL = 3;
    (void)in_sizes; (void)n_in; (void)out_size; (void)ws_size;

    const float* feats   = (const float*)d_in[0];
    const float* pair_in = (const float*)d_in[1];
    const unsigned char* mask1 = (const unsigned char*)d_in[2];
    const unsigned char* maskp = (const unsigned char*)d_in[3];
    // d_in[4] = coords (unused by the forward math)

    // params pytree, dict keys flattened in sorted order
    const float* b_atom_cond = (const float*)d_in[5];
    const float* b_ts        = (const float*)d_in[6];
    const float* bias_b1     = (const float*)d_in[7];
    const float* bias_b2     = (const float*)d_in[8];
    const float* bias_w1     = (const float*)d_in[9];
    const float* bias_w2     = (const float*)d_in[10];
    const float* bk          = (const float*)d_in[11];
    const float* bo          = (const float*)d_in[12];
    const float* bq          = (const float*)d_in[13];
    const float* bv          = (const float*)d_in[14];
    const float* ln_b        = (const float*)d_in[15];
    const float* ln_g        = (const float*)d_in[16];
    const float* pair_bh     = (const float*)d_in[17];
    const float* pair_bw     = (const float*)d_in[18];
    const float* pair_m1     = (const float*)d_in[19];
    const float* pair_m2     = (const float*)d_in[20];
    const float* pair_mb1    = (const float*)d_in[21];
    const float* pair_mb2    = (const float*)d_in[22];
    const float* pair_wh     = (const float*)d_in[23];
    const float* pair_ww     = (const float*)d_in[24];
    const float* pos_embed   = (const float*)d_in[25];
    const float* t_b1        = (const float*)d_in[26];
    const float* t_b2        = (const float*)d_in[27];
    const float* t_w1        = (const float*)d_in[28];
    const float* t_w2        = (const float*)d_in[29];
    const float* tp_b1       = (const float*)d_in[30];
    const float* tp_b2       = (const float*)d_in[31];
    const float* tp_w1       = (const float*)d_in[32];
    const float* tp_w2       = (const float*)d_in[33];
    const float* w_atom_cond = (const float*)d_in[34];
    const float* w_ts        = (const float*)d_in[35];
    const float* wk          = (const float*)d_in[36];
    const float* wo          = (const float*)d_in[37];
    const float* wq          = (const float*)d_in[38];
    const float* wv          = (const float*)d_in[39];

    // workspace layout (f32)
    float* ws = (float*)d_ws;
    float* X  = ws;                 // [N,D]
    float* H0 = X  + N * D;         // LN out
    float* H1 = H0 + N * D;         // FF hidden [N,FF]
    float* Q  = H1 + N * FF;
    float* Kb = Q  + N * D;
    float* Vb = Kb + N * D;
    float* O  = Vb + N * D;
    float* S  = O  + N * D;         // [H,N,N]
    float* PH = S  + (long long)H * N * N;
    float* PW = PH + N * 16;
    float* PF = PW + N * 16;        // [N*N,16]
    float* PB = PF + (long long)N * N * 16;  // pair bias [N,N]

    // 1. atom_cond = feats @ w_atom_cond + b + pos_embed  (pos_embed as residual)
    launch_gemm(stream, feats, TS, 0, w_atom_cond, D, 0, 0,
                X, D, 0, b_atom_cond, pos_embed, D, N, D, TS, 0, 1);

    // 2. ph / pw projections (N=16)
    launch_gemm(stream, X, D, 0, pair_wh, 16, 0, 0, PH, 16, 0, pair_bh, nullptr, 0,
                N, 16, D, 0, 1);
    launch_gemm(stream, X, D, 0, pair_ww, 16, 0, 0, PW, 16, 0, pair_bw, nullptr, 0,
                N, 16, D, 0, 1);

    // 3. pf = relu((pair_in + ph_i + pw_j) @ m1 + mb1) @ m2 + mb2
    pair_feat_kernel<<<(N * N) / 256, 256, 0, stream>>>(
        pair_in, PH, PW, pair_m1, pair_mb1, pair_m2, pair_mb2, PF, N);

    // 4. attention bias from pf (fused WMMA + LDS reduction)
    pair_bias_kernel<<<(N * N) / 16, 256, 0, stream>>>(
        PF, bias_w1, bias_b1, bias_w2, bias_b2, PB);

    const float scale = 0.125f;  // HD^-0.5, HD=64
    for (int l = 0; l < NL; ++l) {
        const float* lw1 = t_w1 + (long long)l * D * FF;
        const float* lw2 = t_w2 + (long long)l * FF * D;
        // Transition: LN -> SiLU MLP -> residual
        layernorm_kernel<<<N, 256, 0, stream>>>(X, ln_g + l * D, ln_b + l * D, H0, D);
        launch_gemm(stream, H0, D, 0, lw1, FF, 0, 0, H1, FF, 0,
                    t_b1 + l * FF, nullptr, 0, N, FF, D, 2, 1);
        launch_gemm(stream, H1, FF, 0, lw2, D, 0, 0, X, D, 0,
                    t_b2 + l * D, X, D, N, D, FF, 0, 1);
        // QKV projections
        launch_gemm(stream, X, D, 0, wq + (long long)l * D * D, D, 0, 0, Q, D, 0,
                    bq + l * D, nullptr, 0, N, D, D, 0, 1);
        launch_gemm(stream, X, D, 0, wk + (long long)l * D * D, D, 0, 0, Kb, D, 0,
                    bk + l * D, nullptr, 0, N, D, D, 0, 1);
        launch_gemm(stream, X, D, 0, wv + (long long)l * D * D, D, 0, 0, Vb, D, 0,
                    bv + l * D, nullptr, 0, N, D, D, 0, 1);
        // S = Q @ K^T per head (batched, transB)
        launch_gemm(stream, Q, D, 64, Kb, D, 64, 1, S, N, (long long)N * N,
                    nullptr, nullptr, 0, N, N, 64, 0, H);
        // softmax(scale*S + bias, mask)
        softmax_bias_kernel<<<dim3(N, H), 256, 0, stream>>>(S, PB, mask1, maskp, N, scale);
        // O = P @ V per head
        launch_gemm(stream, S, N, (long long)N * N, Vb, D, 64, 0, O, D, 64,
                    nullptr, nullptr, 0, N, 64, N, 0, H);
        // x = O @ wo + bo
        launch_gemm(stream, O, D, 0, wo + (long long)l * D * D, D, 0, 0, X, D, 0,
                    bo + l * D, nullptr, 0, N, D, D, 0, 1);
    }

    // outputs: [x | token_single | token_pair]
    float* out  = (float*)d_out;
    float* oTS  = out + N * D;
    float* oTP  = oTS + N * TS;
    copy_kernel<<<(N * D + 255) / 256, 256, 0, stream>>>(X, out, N * D);
    launch_gemm(stream, X, D, 0, w_ts, TS, 0, 0, oTS, TS, 0,
                b_ts, nullptr, 0, N, TS, D, 0, 1);
    token_pair_kernel<<<(N * N) / 16, 256, 0, stream>>>(
        PF, tp_w1, tp_b1, tp_w2, tp_b2, oTP);
}
